// SelfAttention_27616639713576
// MI455X (gfx1250) — matrix operand order repaired and checked by reading, not compiled, not run on hardware
//
#include <hip/hip_runtime.h>
#include <hip/hip_bf16.h>

typedef __attribute__((ext_vector_type(16))) _Float16 v16h;
typedef __attribute__((ext_vector_type(8)))  float    v8f;

#define L2E 1.4426950408889634f

union V16U { v16h h; uint4 u[2]; };
union H8U  { uint4 u; _Float16 h[8]; };

static __device__ __forceinline__ v8f wmma16(v16h a, v16h b, v8f c) {
  // D = A(16x32 f16) * B(32x16 f16) + C(16x16 f32)
  return __builtin_amdgcn_wmma_f32_16x16x32_f16(false, a, false, b, (short)0, c, false, false);
}

// B fragment chunking: halves p[0..7] -> VGPR0..3, p[16..23] -> VGPR4..7.
static __device__ __forceinline__ v16h ldB16(const _Float16* p) {
  V16U t;
  t.u[0] = *(const uint4*)(p);
  t.u[1] = *(const uint4*)(p + 16);
  return t.h;
}

// A fragment chunking: halves p[0..7] -> VGPR0..3, p[16..23] -> VGPR4..7.
static __device__ __forceinline__ v16h ldA16(const _Float16* p) {
  V16U t;
  t.u[0] = *(const uint4*)(p);
  t.u[1] = *(const uint4*)(p + 16);
  return t.h;
}

// A fragment from fp32 source with convert.
static __device__ __forceinline__ v16h ldA32cvt(const float* p) {
  float4 x0 = *(const float4*)(p);
  float4 x1 = *(const float4*)(p + 4);
  float4 y0 = *(const float4*)(p + 16);
  float4 y1 = *(const float4*)(p + 20);
  v16h a;
  a[0] = (_Float16)x0.x; a[1] = (_Float16)x0.y; a[2] = (_Float16)x0.z; a[3] = (_Float16)x0.w;
  a[4] = (_Float16)x1.x; a[5] = (_Float16)x1.y; a[6] = (_Float16)x1.z; a[7] = (_Float16)x1.w;
  a[8] = (_Float16)y0.x; a[9] = (_Float16)y0.y; a[10] = (_Float16)y0.z; a[11] = (_Float16)y0.w;
  a[12] = (_Float16)y1.x; a[13] = (_Float16)y1.y; a[14] = (_Float16)y1.z; a[15] = (_Float16)y1.w;
  return a;
}

// Fused DPP 16-lane reductions: max/add with the permute applied to src0 (1 VALU each).
static __device__ __forceinline__ float dpp16_max(float x) {
  float r;
  asm("v_max_num_f32_dpp %0, %1, %1 quad_perm:[1,0,3,2] row_mask:0xf bank_mask:0xf bound_ctrl:1"
      : "=v"(r) : "v"(x));
  asm("v_max_num_f32_dpp %0, %1, %1 quad_perm:[2,3,0,1] row_mask:0xf bank_mask:0xf bound_ctrl:1"
      : "=v"(x) : "v"(r));
  asm("v_max_num_f32_dpp %0, %1, %1 row_half_mirror row_mask:0xf bank_mask:0xf bound_ctrl:1"
      : "=v"(r) : "v"(x));
  asm("v_max_num_f32_dpp %0, %1, %1 row_mirror row_mask:0xf bank_mask:0xf bound_ctrl:1"
      : "=v"(x) : "v"(r));
  return x;
}
static __device__ __forceinline__ float dpp16_sum(float x) {
  float r;
  asm("v_add_f32_dpp %0, %1, %1 quad_perm:[1,0,3,2] row_mask:0xf bank_mask:0xf bound_ctrl:1"
      : "=v"(r) : "v"(x));
  asm("v_add_f32_dpp %0, %1, %1 quad_perm:[2,3,0,1] row_mask:0xf bank_mask:0xf bound_ctrl:1"
      : "=v"(x) : "v"(r));
  asm("v_add_f32_dpp %0, %1, %1 row_half_mirror row_mask:0xf bank_mask:0xf bound_ctrl:1"
      : "=v"(r) : "v"(x));
  asm("v_add_f32_dpp %0, %1, %1 row_mirror row_mask:0xf bank_mask:0xf bound_ctrl:1"
      : "=v"(x) : "v"(r));
  return x;
}

// Async DMA: global -> LDS, 16 bytes per lane, tracked by ASYNCcnt.
static __device__ __forceinline__ void async_ld16(unsigned lds_byte_addr, const _Float16* g) {
  asm volatile("global_load_async_to_lds_b128 %0, %1, off"
               :: "v"(lds_byte_addr), "v"(g) : "memory");
}

// ---- Kernel 1: convert + transpose weights to f16 [f][e]; fold 1/sqrt(D) into Wq ----
__global__ void __launch_bounds__(256) convert_w(
    const float* __restrict__ Wq, const float* __restrict__ Wk,
    const float* __restrict__ Wv, const float* __restrict__ Wo,
    _Float16* __restrict__ Wqt, _Float16* __restrict__ Wkt,
    _Float16* __restrict__ Wvt, _Float16* __restrict__ Wot) {
  int idx = blockIdx.x * 256 + threadIdx.x;   // 512*512 elements
  int i = idx >> 9, j = idx & 511;
  int tj = j * 512 + i;
  Wqt[tj] = (_Float16)(Wq[idx] * 0.125f);     // fold 1/sqrt(64)
  Wkt[tj] = (_Float16)Wk[idx];
  Wvt[tj] = (_Float16)Wv[idx];
  Wot[tj] = (_Float16)Wo[idx];
}

// ---- Kernel 2: projection GEMM. One wave: 16 rows x 128 cols (8 WMMA col-tiles),
// A fragment loaded once per k-step and reused for all 8 tiles.
// MODE 0: natural [row*512+col].  MODE 1: V transposed -> Vt[n][h][d][s].
template <int MODE>
__global__ void __launch_bounds__(256) proj_gemm(
    const float* __restrict__ A, const _Float16* __restrict__ Wt,
    _Float16* __restrict__ C) {
  const int tid = threadIdx.x, lane = tid & 31, w = tid >> 5;
  const int lq = lane & 15, hf = lane >> 4;
  const int tile = blockIdx.x * 8 + w;        // 4096 waves total
  const int tm = tile >> 2;                   // 0..1023 row tile
  const int cg = tile & 3;                    // col group: 128 cols
  const int row0 = tm << 4;

  const float* arow = A + (size_t)(row0 + lq) * 512;
  const _Float16* brow = Wt + (size_t)(cg * 128 + lq) * 512;
  v8f c[8] = {};
  for (int k = 0; k < 512; k += 32) {
    v16h a = ldA32cvt(arow + k + hf * 8);
#pragma unroll
    for (int j = 0; j < 8; ++j) {
      v16h b = ldB16(brow + (size_t)j * 8192 + k + hf * 8);
      c[j] = wmma16(a, b, c[j]);
    }
  }
#pragma unroll
  for (int j = 0; j < 8; ++j) {
    const int col = cg * 128 + j * 16 + lq;
    if (MODE == 0) {
#pragma unroll
      for (int g = 0; g < 8; ++g)
        C[(size_t)(row0 + (hf << 3) + g) * 512 + col] = (_Float16)c[j][g];
    } else {
      const int n = row0 >> 11;
      const int s = (row0 & 2047) + (hf << 3);
      const int head = col >> 6, d = col & 63;
      H8U t;
#pragma unroll
      for (int g = 0; g < 8; ++g) t.h[g] = (_Float16)c[j][g];
      *(uint4*)(C + (size_t)((n * 8 + head) * 64 + d) * 2048 + s) = t.u;
    }
  }
}

// ---- Kernel 3: flash attention. Block = one (n,h), 8 waves = 8 consecutive q-tiles.
// K/V 32-key tiles DMA'd into LDS (async-to-LDS), triple buffered, pipelined 1 deep.
__global__ void __launch_bounds__(256) attn_kernel(
    const _Float16* __restrict__ Qh, const _Float16* __restrict__ Kh,
    const _Float16* __restrict__ Vt, const int* __restrict__ mask,
    _Float16* __restrict__ ctx) {
  __shared__ _Float16 ksm[3][32 * 64];        // [key][d], 4KB per buffer
  __shared__ _Float16 vsm[3][64 * 32];        // [d][key], 4KB per buffer
  __shared__ _Float16 pt[8][16][40];          // per-wave P tile, padded rows

  const int tid = threadIdx.x, lane = tid & 31, w = tid >> 5;
  const int lq = lane & 15, hf = lane >> 4;
  const int blk = blockIdx.x;                 // 1024 blocks
  const int qg = blk & 15;
  const int h = (blk >> 4) & 7;
  const int n = blk >> 7;
  const int qt = qg * 8 + w;                  // q-tile of this wave

  // DMA mapping: one 16B chunk per thread per tile
  const int kkey = tid >> 3, kch = tid & 7;   // K: 32 rows x 8 chunks
  const int vd = tid >> 2, vch = tid & 3;     // V: 64 rows x 4 chunks
  const _Float16* kgp = Kh + (size_t)(n * 2048 + kkey) * 512 + h * 64 + kch * 8;
  const _Float16* vgp = Vt + (size_t)((n * 8 + h) * 64 + vd) * 2048 + vch * 8;
  // LDS byte addresses (low 32 bits of a generic LDS address = LDS byte offset)
  const unsigned klds = (unsigned)(uintptr_t)&ksm[0][kkey * 64 + kch * 8];
  const unsigned vlds = (unsigned)(uintptr_t)&vsm[0][vd * 32 + vch * 8];
  const int* mrow = mask + n * 2048;

  const _Float16* qb = Qh + (size_t)(n * 2048 + qt * 16) * 512 + h * 64;
  v16h aq0 = ldA16(qb + (size_t)lq * 512 + hf * 8);        // d 0..31 block
  v16h aq1 = ldA16(qb + (size_t)lq * 512 + 32 + hf * 8);   // d 32..63 block

  v8f o0 = {}, o1 = {}, o2 = {}, o3 = {};
  float m[8], l[8];
#pragma unroll
  for (int g = 0; g < 8; ++g) { m[g] = -3.0e38f; l[g] = 0.0f; }

  // prologue: DMA tile 0 into buffer 0
  async_ld16(klds, kgp);
  async_ld16(vlds, vgp);

  for (int kt = 0; kt < 64; ++kt) {
    const int key0 = kt << 5;
    const int buf = kt - (kt / 3) * 3;        // kt % 3
    // issue DMA for the next tile into buffer (kt+1)%3
    {
      const int nx = (kt < 63) ? (kt + 1) : 63;
      const int nb = nx - (nx / 3) * 3;
      async_ld16(klds + nb * 4096, kgp + ((size_t)nx << 5) * 512);
      async_ld16(vlds + nb * 4096, vgp + (nx << 5));
    }
    // <=2 outstanding: the 2 loads of the current tile have landed (in-order)
    asm volatile("s_wait_asynccnt 0x2" ::: "memory");
    __syncthreads();

    // S = Q K^T for 16 q-rows x 32 keys
    v8f c0 = {}, c1 = {};
    {
      const _Float16* p = &ksm[buf][lq * 64 + hf * 8];
      c0 = wmma16(aq0, ldB16(p), c0);
      c0 = wmma16(aq1, ldB16(p + 32), c0);
    }
    {
      const _Float16* p = &ksm[buf][(16 + lq) * 64 + hf * 8];
      c1 = wmma16(aq0, ldB16(p), c1);
      c1 = wmma16(aq1, ldB16(p + 32), c1);
    }
    const int mk0 = mrow[key0 + lq];
    const int mk1 = mrow[key0 + 16 + lq];
#pragma unroll
    for (int g = 0; g < 8; ++g) {
      if (!mk0) c0[g] = -1e20f;
      if (!mk1) c1[g] = -1e20f;
    }
    // online softmax: each row lives in one 16-lane half, same VGPR index
    float p0[8], p1[8];
#pragma unroll
    for (int g = 0; g < 8; ++g) {
      float t = dpp16_max(fmaxf(c0[g], c1[g]));
      float mn = fmaxf(m[g], t);
      float al = exp2f((m[g] - mn) * L2E);
      p0[g] = exp2f((c0[g] - mn) * L2E);
      p1[g] = exp2f((c1[g] - mn) * L2E);
      float rs = dpp16_sum(p0[g] + p1[g]);
      l[g] = l[g] * al + rs;
      m[g] = mn;
      o0[g] *= al; o1[g] *= al; o2[g] *= al; o3[g] *= al;
    }
    // C-layout -> A-layout transpose of P via per-wave LDS tile
#pragma unroll
    for (int g = 0; g < 8; ++g) {
      const int r = g + (hf << 3);
      pt[w][r][lq] = (_Float16)p0[g];
      pt[w][r][16 + lq] = (_Float16)p1[g];
    }
    asm volatile("s_wait_dscnt 0x0" ::: "memory");
    v16h ap = ldA16(&pt[w][lq][hf * 8]);      // A frag of P(16x32)
    // PV from LDS: vsm[d][key]
    const _Float16* vbl = &vsm[buf][hf * 8];
    o0 = wmma16(ap, ldB16(vbl + (lq) * 32), o0);
    o1 = wmma16(ap, ldB16(vbl + (16 + lq) * 32), o1);
    o2 = wmma16(ap, ldB16(vbl + (32 + lq) * 32), o2);
    o3 = wmma16(ap, ldB16(vbl + (48 + lq) * 32), o3);
  }
  _Float16* cb = ctx + (size_t)(n * 2048 + qt * 16) * 512 + h * 64;
#pragma unroll
  for (int g = 0; g < 8; ++g) {
    const int r = g + (hf << 3);
    const float rl = __builtin_amdgcn_rcpf(l[g]);
    cb[(size_t)r * 512 + lq]      = (_Float16)(o0[g] * rl);
    cb[(size_t)r * 512 + 16 + lq] = (_Float16)(o1[g] * rl);
    cb[(size_t)r * 512 + 32 + lq] = (_Float16)(o2[g] * rl);
    cb[(size_t)r * 512 + 48 + lq] = (_Float16)(o3[g] * rl);
  }
}

// ---- Kernel 4: output projection. out = ctx(f16) * Wot^T + bo, fp32 out.
// Same 8-col-tile blocking as proj_gemm.
__global__ void __launch_bounds__(256) out_gemm(
    const _Float16* __restrict__ Ax, const _Float16* __restrict__ Wt,
    const float* __restrict__ bo, float* __restrict__ out) {
  const int tid = threadIdx.x, lane = tid & 31, w = tid >> 5;
  const int lq = lane & 15, hf = lane >> 4;
  const int tile = blockIdx.x * 8 + w;
  const int tm = tile >> 2;
  const int cg = tile & 3;
  const int row0 = tm << 4;

  const _Float16* arow = Ax + (size_t)(row0 + lq) * 512;
  const _Float16* brow = Wt + (size_t)(cg * 128 + lq) * 512;
  v8f c[8] = {};
  for (int k = 0; k < 512; k += 32) {
    v16h a = ldA16(arow + k + hf * 8);
#pragma unroll
    for (int j = 0; j < 8; ++j) {
      v16h b = ldB16(brow + (size_t)j * 8192 + k + hf * 8);
      c[j] = wmma16(a, b, c[j]);
    }
  }
#pragma unroll
  for (int j = 0; j < 8; ++j) {
    const int col = cg * 128 + j * 16 + lq;
    const float bias = bo[col];
#pragma unroll
    for (int g = 0; g < 8; ++g)
      out[(size_t)(row0 + (hf << 3) + g) * 512 + col] = c[j][g] + bias;
  }
}

extern "C" void kernel_launch(void* const* d_in, const int* in_sizes, int n_in,
                              void* d_out, int out_size, void* d_ws, size_t ws_size,
                              hipStream_t stream) {
  (void)in_sizes; (void)n_in; (void)out_size; (void)ws_size;
  const float* values = (const float*)d_in[0];
  const float* keys   = (const float*)d_in[1];
  const float* query  = (const float*)d_in[2];
  const int*   mask   = (const int*)d_in[3];
  const float* Wv = (const float*)d_in[4];
  const float* Wk = (const float*)d_in[5];
  const float* Wq = (const float*)d_in[6];
  const float* Wo = (const float*)d_in[7];
  const float* bo = (const float*)d_in[8];
  float* out = (float*)d_out;

  char* ws = (char*)d_ws;
  const size_t WT  = (size_t)512 * 512 * 2;    // 0.5 MB per weight
  const size_t BIG = (size_t)16384 * 512 * 2;  // 16 MB per activation
  _Float16* Wqt = (_Float16*)(ws + 0 * WT);
  _Float16* Wkt = (_Float16*)(ws + 1 * WT);
  _Float16* Wvt = (_Float16*)(ws + 2 * WT);
  _Float16* Wot = (_Float16*)(ws + 3 * WT);
  _Float16* Qh  = (_Float16*)(ws + 4 * WT);
  _Float16* Kh  = (_Float16*)(ws + 4 * WT + 1 * BIG);
  _Float16* Vt  = (_Float16*)(ws + 4 * WT + 2 * BIG);
  _Float16* Ctx = (_Float16*)(ws + 4 * WT + 3 * BIG);

  convert_w<<<1024, 256, 0, stream>>>(Wq, Wk, Wv, Wo, Wqt, Wkt, Wvt, Wot);
  proj_gemm<0><<<512, 256, 0, stream>>>(query,  Wqt, Qh);
  proj_gemm<0><<<512, 256, 0, stream>>>(keys,   Wkt, Kh);
  proj_gemm<1><<<512, 256, 0, stream>>>(values, Wvt, Vt);
  attn_kernel<<<1024, 256, 0, stream>>>(Qh, Kh, Vt, mask, Ctx);
  out_gemm<<<512, 256, 0, stream>>>(Ctx, Wot, bo, out);
}
